// KNN_69217692942515
// MI455X (gfx1250) — matrix operand order repaired and checked by reading, not compiled, not run on hardware
//
#include <hip/hip_runtime.h>
#include <hip/hip_bf16.h>

typedef __attribute__((ext_vector_type(16))) _Float16 v16h;
typedef __attribute__((ext_vector_type(8)))  _Float16 v8h;
typedef __attribute__((ext_vector_type(8)))  float    v8f;

#define N_ROWS 12288
#define DIM    256
#define KNN_K  32

// ---------------- Phase 1: L2-normalize rows, fp32 -> f16 ----------------
// One wave (32 lanes) per row; lane handles 8 consecutive elements.
__global__ void __launch_bounds__(256) l2norm_f16(const float* __restrict__ x,
                                                  _Float16* __restrict__ xn) {
    const int lane = threadIdx.x & 31;
    const int row  = blockIdx.x * 8 + (threadIdx.x >> 5);
    const float* rp = x + (size_t)row * DIM + lane * 8;

    float v[8];
#pragma unroll
    for (int j = 0; j < 8; ++j) v[j] = rp[j];

    float s = 0.0f;
#pragma unroll
    for (int j = 0; j < 8; ++j) s += v[j] * v[j];
#pragma unroll
    for (int off = 16; off > 0; off >>= 1) s += __shfl_xor(s, off, 32);

    const float scale = 1.0f / fmaxf(sqrtf(s), 1e-12f);

    v8h o;
#pragma unroll
    for (int j = 0; j < 8; ++j) o[j] = (_Float16)(v[j] * scale);
    *(v8h*)(xn + (size_t)row * DIM + lane * 8) = o;
}

// ---------------- Phase 2: adj = Xn * Xn^T via f16 WMMA ----------------
// Block = 256 threads = 8 waves arranged 4(row) x 2(col) -> 256x128 output block.
// Each wave computes a 4x4 grid of 16x16 tiles (64x64 output):
//   16 WMMAs per K-step against 16 b128 loads -> 1.0 wmma/load.
// f16 fragment layouts (ISA 7.12.2):
//   A 16x32: lane L (m=L&15, h=L>>4) holds row m, K = h*8+[0..7] and 16+h*8+[0..7]
//   B 32x16: lane L (n=L&15, h=L>>4) holds col n, K = h*8+[0..7] and 16+h*8+[0..7]
//   C 16x16 f32: elem r -> row (h*8 + r), col (L&15)
// Since B columns are Xn rows, both fragments are contiguous row chunks.
__global__ void __launch_bounds__(256) cosine_gemm_wmma(const _Float16* __restrict__ xn,
                                                        float* __restrict__ out) {
    const int lane = threadIdx.x & 31;
    const int wid  = threadIdx.x >> 5;
    const int wr   = wid >> 1;       // 0..3
    const int wc   = wid & 1;        // 0..1
    const int m    = lane & 15;
    const int h    = lane >> 4;

    const int i0 = blockIdx.y * 256 + wr * 64;   // 4 tiles of rows
    const int j0 = blockIdx.x * 128 + wc * 64;   // 4 tiles of cols

    const _Float16* Abase[4];
    const _Float16* Bbase[4];
#pragma unroll
    for (int a = 0; a < 4; ++a) Abase[a] = xn + (size_t)(i0 + a * 16 + m) * DIM;
#pragma unroll
    for (int b = 0; b < 4; ++b) Bbase[b] = xn + (size_t)(j0 + b * 16 + m) * DIM;

    v8f acc[4][4];
#pragma unroll
    for (int a = 0; a < 4; ++a)
#pragma unroll
        for (int b = 0; b < 4; ++b) acc[a][b] = (v8f){};

    for (int k0 = 0; k0 < DIM; k0 += 32) {
        v16h afr[4], bfr[4];
#pragma unroll
        for (int a = 0; a < 4; ++a) {
            v8h p = *(const v8h*)(Abase[a] + k0 + h * 8);        // K = h*8 .. +8
            v8h q = *(const v8h*)(Abase[a] + k0 + 16 + h * 8);   // K = 16+h*8 .. +8
            afr[a] = __builtin_shufflevector(p, q, 0, 1, 2, 3, 4, 5, 6, 7,
                                                   8, 9, 10, 11, 12, 13, 14, 15);
        }
#pragma unroll
        for (int b = 0; b < 4; ++b) {
            v8h p = *(const v8h*)(Bbase[b] + k0 + h * 8);        // K = h*8 .. +8
            v8h q = *(const v8h*)(Bbase[b] + k0 + 16 + h * 8);   // K = 16+h*8 .. +8
            bfr[b] = __builtin_shufflevector(p, q, 0, 1, 2, 3, 4, 5, 6, 7,
                                                   8, 9, 10, 11, 12, 13, 14, 15);
        }

#pragma unroll
        for (int a = 0; a < 4; ++a)
#pragma unroll
            for (int b = 0; b < 4; ++b)
                acc[a][b] = __builtin_amdgcn_wmma_f32_16x16x32_f16(
                    false, afr[a], false, bfr[b], (short)0, acc[a][b], false, false);
    }

#pragma unroll
    for (int a = 0; a < 4; ++a)
#pragma unroll
        for (int b = 0; b < 4; ++b) {
            const int col = j0 + b * 16 + m;
#pragma unroll
            for (int r = 0; r < 8; ++r) {
                const int row = i0 + a * 16 + h * 8 + r;
                out[(size_t)row * N_ROWS + col] = acc[a][b][r];
            }
        }
}

// ---------------- Phase 3: per-row top-32 threshold mask (in place) ----------------
// One block per row. Exact 32nd-largest via binary search on monotonic uint keys.
// 12288 = 256 threads * 48 keys, keys kept in VGPRs; map is invertible so no
// second value array is needed.
__global__ void __launch_bounds__(256) topk_mask(float* __restrict__ adj) {
    float* rp = adj + (size_t)blockIdx.x * N_ROWS;
    const int tid = threadIdx.x;

    unsigned key[48];
#pragma unroll
    for (int j = 0; j < 48; ++j) {
        unsigned bits = __float_as_uint(rp[tid + j * 256]);
        key[j] = bits ^ (unsigned)(((int)bits >> 31) | 0x80000000);
    }

    __shared__ unsigned cnt;
    unsigned lo = 0u, hi = 0xFFFFFFFFu;
    while (lo < hi) {
        // upper midpoint in (lo, hi], overflow-safe
        const unsigned mid = lo + ((hi - lo) >> 1) + 1u;
        if (tid == 0) cnt = 0u;
        __syncthreads();

        unsigned local = 0u;
#pragma unroll
        for (int j = 0; j < 48; ++j) local += (key[j] >= mid) ? 1u : 0u;
#pragma unroll
        for (int off = 16; off > 0; off >>= 1) local += __shfl_xor(local, off, 32);
        if ((tid & 31) == 0) atomicAdd(&cnt, local);
        __syncthreads();

        const unsigned total = cnt;   // uniform across block
        if (total >= (unsigned)KNN_K) lo = mid; else hi = mid - 1u;
        __syncthreads();              // protect cnt reset in next iteration
    }
    const unsigned thresh = lo;       // key of the K-th largest element

#pragma unroll
    for (int j = 0; j < 48; ++j) {
        float v = 0.0f;
        if (key[j] >= thresh) {
            const unsigned u = key[j];
            const unsigned bits = (u & 0x80000000u) ? (u ^ 0x80000000u) : ~u;
            v = __uint_as_float(bits);
        }
        rp[tid + j * 256] = v;
    }
}

extern "C" void kernel_launch(void* const* d_in, const int* in_sizes, int n_in,
                              void* d_out, int out_size, void* d_ws, size_t ws_size,
                              hipStream_t stream) {
    const float* x   = (const float*)d_in[0];
    float*       out = (float*)d_out;
    _Float16*    xn  = (_Float16*)d_ws;   // 12288*256*2 = 6.3 MB

    l2norm_f16<<<N_ROWS / 8, 256, 0, stream>>>(x, xn);

    dim3 grid(N_ROWS / 128, N_ROWS / 256);   // 96 x 48
    cosine_gemm_wmma<<<grid, 256, 0, stream>>>(xn, out);

    topk_mask<<<N_ROWS, 256, 0, stream>>>(out);
}
